// SelfAttnHead_26645977105019
// MI455X (gfx1250) — matrix operand order repaired and checked by reading, not compiled, not run on hardware
//
#include <hip/hip_runtime.h>

#define BB 16
#define TT 2048
#define EE 1024
#define HH 128

typedef __attribute__((ext_vector_type(16))) _Float16 v16h;
typedef __attribute__((ext_vector_type(8)))  float    v8f;

union V16H { v16h v; uint4 u[2]; };

static __device__ __forceinline__ v8f wmma_f16(v16h a, v16h b, v8f c) {
  // D = A(16x32 f16) * B(32x16 f16) + C(16x16 f32)
  return __builtin_amdgcn_wmma_f32_16x16x32_f16(false, a, false, b, (short)0, c, false, false);
}

static __device__ __forceinline__ void load_b_frag(
    const _Float16* __restrict__ Wpk, int lane, int kc, int ct,
    V16H& bq_, V16H& bv_) {
  const _Float16* pb = Wpk + ((size_t)(ct * (EE / 32) + kc) << 9) + lane * 16;
  const _Float16* pv = pb + (size_t)EE * HH;
  bq_.u[0] = *(const uint4*)pb;  bq_.u[1] = *(const uint4*)(pb + 8);
  bv_.u[0] = *(const uint4*)pv;  bv_.u[1] = *(const uint4*)(pv + 8);
}

// ---------------------------------------------------------------------------
// Pack W (ExH, row-major f32) into per-wave B-fragment order, f16.
//   col  = ct*16 + (lane&15)
//   krow = kc*32 + (lane>>4)*16 + j
// ---------------------------------------------------------------------------
__global__ void __launch_bounds__(256) pack_w_kernel(
    const float* __restrict__ Wq, const float* __restrict__ Wv,
    _Float16* __restrict__ Wpk)
{
  int idx = blockIdx.x * 256 + threadIdx.x;     // exact grid, no bounds check
  const int per = EE * HH;
  int which = idx / per;
  int s = idx - which * per;
  int frag = s >> 9;
  int t = s & 511;
  int lane = t >> 4, j = t & 15;
  int ct = frag / (EE / 32);
  int kc = frag - ct * (EE / 32);
  int col = ct * 16 + (lane & 15);
  int krow = kc * 32 + (lane >> 4) * 16 + j;
  const float* W = which ? Wv : Wq;
  Wpk[idx] = (_Float16)W[krow * HH + col];
}

// ---------------------------------------------------------------------------
// Projection: Q = (x@Wq + bq) * H^-0.25  (scale folded; reused as q and k)
//             V = (x@Wv + bv)
// 8 waves/block, 16 rows/wave. Software-pipelined: B-fragment double buffer,
// A-fragment ping-pong in *named* registers (kc unrolled by 2).
// ---------------------------------------------------------------------------
__global__ void __launch_bounds__(256) proj_kernel(
    const float* __restrict__ x, const _Float16* __restrict__ Wpk,
    const float* __restrict__ bq, const float* __restrict__ bv,
    _Float16* __restrict__ Qs, _Float16* __restrict__ Vs)
{
  const int wave = threadIdx.x >> 5, lane = threadIdx.x & 31;
  const int h = lane >> 4, n = lane & 15;
  const int rowbase = blockIdx.x * 128 + wave * 16;
  const float* xrow = x + (size_t)(rowbase + n) * EE;

  v8f qacc[8] = {};
  v8f vacc[8] = {};

  float4 xf0, xf1, xf2, xf3;
  V16H aE, aO;                 // named A fragments (no dynamic indexing)
  V16H bqf[2], bvf[2];

#define ISSUE_X(kcn) do {                                        \
    const float* p0_ = xrow + (kcn) * 32 + h * 16;               \
    xf0 = *(const float4*)(p0_);                                 \
    xf1 = *(const float4*)(p0_ + 4);                             \
    xf2 = *(const float4*)(p0_ + 8);                             \
    xf3 = *(const float4*)(p0_ + 12);                            \
  } while (0)

#define CVT_A(dst) do {                                          \
    (dst).v[0]  = (_Float16)xf0.x; (dst).v[1]  = (_Float16)xf0.y;\
    (dst).v[2]  = (_Float16)xf0.z; (dst).v[3]  = (_Float16)xf0.w;\
    (dst).v[4]  = (_Float16)xf1.x; (dst).v[5]  = (_Float16)xf1.y;\
    (dst).v[6]  = (_Float16)xf1.z; (dst).v[7]  = (_Float16)xf1.w;\
    (dst).v[8]  = (_Float16)xf2.x; (dst).v[9]  = (_Float16)xf2.y;\
    (dst).v[10] = (_Float16)xf2.z; (dst).v[11] = (_Float16)xf2.w;\
    (dst).v[12] = (_Float16)xf3.x; (dst).v[13] = (_Float16)xf3.y;\
    (dst).v[14] = (_Float16)xf3.z; (dst).v[15] = (_Float16)xf3.w;\
  } while (0)

  ISSUE_X(0);
  CVT_A(aE);
  load_b_frag(Wpk, lane, 0, 0, bqf[0], bvf[0]);

  for (int kc2 = 0; kc2 < EE / 32; kc2 += 2) {
    // ---- even kc = kc2 : compute with aE, prepare aO(kc2+1) ----
    #pragma unroll
    for (int ct = 0; ct < 8; ++ct) {
      const int s = ct & 1, ns = s ^ 1;
      if (ct < 7)
        load_b_frag(Wpk, lane, kc2, ct + 1, bqf[ns], bvf[ns]);
      else
        load_b_frag(Wpk, lane, kc2 + 1, 0, bqf[ns], bvf[ns]);
      if (ct == 0) ISSUE_X(kc2 + 1);
      if (ct == 7) CVT_A(aO);
      qacc[ct] = wmma_f16(aE.v, bqf[s].v, qacc[ct]);
      vacc[ct] = wmma_f16(aE.v, bvf[s].v, vacc[ct]);
    }
    // ---- odd kc = kc2+1 : compute with aO, prepare aE(kc2+2) ----
    #pragma unroll
    for (int ct = 0; ct < 8; ++ct) {
      const int s = ct & 1, ns = s ^ 1;
      if (ct < 7)
        load_b_frag(Wpk, lane, kc2 + 1, ct + 1, bqf[ns], bvf[ns]);
      else if (kc2 + 2 < EE / 32)
        load_b_frag(Wpk, lane, kc2 + 2, 0, bqf[ns], bvf[ns]);
      if (ct == 0 && kc2 + 2 < EE / 32) ISSUE_X(kc2 + 2);
      if (ct == 7 && kc2 + 2 < EE / 32) CVT_A(aE);
      qacc[ct] = wmma_f16(aO.v, bqf[s].v, qacc[ct]);
      vacc[ct] = wmma_f16(aO.v, bvf[s].v, vacc[ct]);
    }
  }
#undef ISSUE_X
#undef CVT_A

  const float ssc = 0.29730177875068026f;   // 128^-0.25
  #pragma unroll
  for (int ct = 0; ct < 8; ++ct) {
    const float bqs = bq[ct * 16 + n];
    const float bvs = bv[ct * 16 + n];
    #pragma unroll
    for (int r = 0; r < 8; ++r) {
      const int row = rowbase + 8 * h + r;          // D layout: m = r + 8*(l>>4)
      const size_t o = (size_t)row * HH + ct * 16 + n;
      Qs[o] = (_Float16)((qacc[ct][r] + bqs) * ssc);
      Vs[o] = (_Float16)(vacc[ct][r] + bvs);
    }
  }
}

// ---------------------------------------------------------------------------
// Transpose V[b][T][H] -> Vt[b][H][T] (f16)
// ---------------------------------------------------------------------------
__global__ void __launch_bounds__(256) transpose_v_kernel(
    const _Float16* __restrict__ Vs, _Float16* __restrict__ Vt)
{
  __shared__ _Float16 tile[32][33];
  const int b = blockIdx.z;
  const int t0 = blockIdx.x * 32, h0 = blockIdx.y * 32;
  const int tx = threadIdx.x & 31, ty = threadIdx.x >> 5;   // 32 x 8
  const _Float16* src = Vs + ((size_t)b * TT + t0) * HH + h0;
  #pragma unroll
  for (int i = 0; i < 32; i += 8)
    tile[ty + i][tx] = src[(size_t)(ty + i) * HH + tx];
  __syncthreads();
  _Float16* dst = Vt + ((size_t)b * HH + h0) * TT + t0;
  #pragma unroll
  for (int i = 0; i < 32; i += 8)
    dst[(size_t)(ty + i) * TT + tx] = tile[tx][ty + i];
}

// ---------------------------------------------------------------------------
// Fused causal flash attention, 64-key blocks. 8 waves/block, 16 q-rows/wave.
// k == q (reference reuses the q projection). Scale folded into Qs.
// ---------------------------------------------------------------------------
__global__ void __launch_bounds__(256) attn_kernel(
    const _Float16* __restrict__ Qs, const _Float16* __restrict__ Vs,
    const _Float16* __restrict__ Vt, const int* __restrict__ msk,
    float* __restrict__ out)
{
  __shared__ _Float16 pstage[8][1024];   // 16x64 f16 P-tile per wave (2 KB)
  const int wave = threadIdx.x >> 5, lane = threadIdx.x & 31;
  const int h = lane >> 4, n = lane & 15;
  const int b = blockIdx.y;
  const int q0 = blockIdx.x * 128 + wave * 16;
  const _Float16* Qb  = Qs + (size_t)b * TT * HH;
  const _Float16* Vtb = Vt + (size_t)b * HH * TT;
  const int* mb = msk + (size_t)b * TT;

  // Persistent Q A-fragments (16 rows x 128 dims = 4 chunks of K=32)
  V16H afrag[4];
  {
    const _Float16* qrow = Qb + (size_t)(q0 + n) * HH;
    #pragma unroll
    for (int kc = 0; kc < 4; ++kc) {
      const _Float16* p0 = qrow + kc * 32 + h * 16;
      afrag[kc].u[0] = *(const uint4*)p0;
      afrag[kc].u[1] = *(const uint4*)(p0 + 8);
    }
  }

  float mrun[8], lrun[8];
  v8f oacc[8] = {};
  #pragma unroll
  for (int r = 0; r < 8; ++r) { mrun[r] = -__builtin_inff(); lrun[r] = 0.0f; }

  _Float16* st = pstage[wave];
  const int kend = q0 + 15;                  // causal: last key needed
  for (int kb = 0; kb <= kend; kb += 64) {
    // prefetch next key block's K rows (global_prefetch_b8)
    if (kb + 128 <= TT && kb + 64 <= kend) {
      __builtin_prefetch(Qb + (size_t)(kb + 64 + lane) * HH, 0, 3);
      __builtin_prefetch(Qb + (size_t)(kb + 96 + lane) * HH, 0, 3);
    }

    // ---- scores S = Q . K^T for 64 keys (four 16x16 tiles) ----
    v8f s4[4] = {};
    #pragma unroll
    for (int kc = 0; kc < 4; ++kc) {
      #pragma unroll
      for (int j = 0; j < 4; ++j) {
        const _Float16* pk = Qb + (size_t)(kb + j * 16 + n) * HH + kc * 32 + h * 16;
        V16H bf;
        bf.u[0] = *(const uint4*)pk;  bf.u[1] = *(const uint4*)(pk + 8);
        s4[j] = wmma_f16(afrag[kc].v, bf.v, s4[j]);
      }
    }

    // ---- online softmax over the 64-key block ----
    int kidx[4], imv[4];
    #pragma unroll
    for (int j = 0; j < 4; ++j) {
      kidx[j] = kb + j * 16 + n;
      imv[j] = mb[kidx[j]];
    }
    float alpha_r[8];
    #pragma unroll
    for (int r = 0; r < 8; ++r) {
      const int qr = q0 + 8 * h + r;                 // D layout row
      bool vv[4];
      float bm = -__builtin_inff();
      #pragma unroll
      for (int j = 0; j < 4; ++j) {
        vv[j] = (kidx[j] <= qr) && (imv[j] != 0);
        bm = fmaxf(bm, vv[j] ? s4[j][r] : -__builtin_inff());
      }
      #pragma unroll
      for (int off = 8; off >= 1; off >>= 1)
        bm = fmaxf(bm, __shfl_xor(bm, off, 32));     // reduce within 16-lane half
      const float mn = fmaxf(mrun[r], bm);
      const float alpha = (mn == -__builtin_inff()) ? 1.0f : __expf(mrun[r] - mn);
      float ls = 0.0f;
      #pragma unroll
      for (int j = 0; j < 4; ++j) {
        const float e = vv[j] ? __expf(s4[j][r] - mn) : 0.0f;
        ls += e;
        st[(8 * h + r) * 64 + j * 16 + n] = (_Float16)e;  // P: row-major 16x64
      }
      #pragma unroll
      for (int off = 8; off >= 1; off >>= 1)
        ls += __shfl_xor(ls, off, 32);
      lrun[r] = lrun[r] * alpha + ls;
      mrun[r] = mn;
      alpha_r[r] = alpha;
    }
    #pragma unroll
    for (int ct = 0; ct < 8; ++ct)
      #pragma unroll
      for (int r = 0; r < 8; ++r) oacc[ct][r] *= alpha_r[r];

    // same-wave LDS RAW: LDS unit is in-order; wait covers data visibility
    asm volatile("s_wait_dscnt 0" ::: "memory");

    // re-read P as two 16x32 A-fragments (keys 0..31 / 32..63 of the block)
    V16H pf0, pf1;
    pf0.u[0] = *(const uint4*)&st[n * 64 + h * 16];
    pf0.u[1] = *(const uint4*)&st[n * 64 + 8 + h * 16];
    pf1.u[0] = *(const uint4*)&st[n * 64 + 32 + h * 16];
    pf1.u[1] = *(const uint4*)&st[n * 64 + 40 + h * 16];

    // ---- O += P . V  (8 column tiles of 16 dims, two K=32 chunks) ----
    #pragma unroll
    for (int ct = 0; ct < 8; ++ct) {
      const _Float16* pv0 = Vtb + (size_t)(ct * 16 + n) * TT + kb + h * 16;
      const _Float16* pv1 = pv0 + 32;
      V16H bv0, bv1;
      bv0.u[0] = *(const uint4*)pv0;  bv0.u[1] = *(const uint4*)(pv0 + 8);
      bv1.u[0] = *(const uint4*)pv1;  bv1.u[1] = *(const uint4*)(pv1 + 8);
      oacc[ct] = wmma_f16(pf0.v, bv0.v, oacc[ct]);
      oacc[ct] = wmma_f16(pf1.v, bv1.v, oacc[ct]);
    }
  }

  // ---- epilogue: normalize; degenerate rows emit v[q] (one-hot diagonal) ----
  const _Float16* Vb = Vs + (size_t)b * TT * HH;
  #pragma unroll
  for (int r = 0; r < 8; ++r) {
    const int qr = q0 + 8 * h + r;
    float* orow = out + ((size_t)b * TT + qr) * HH;
    if (lrun[r] > 0.0f) {
      const float inv = 1.0f / lrun[r];
      #pragma unroll
      for (int ct = 0; ct < 8; ++ct)
        orow[ct * 16 + n] = oacc[ct][r] * inv;
    } else {
      const _Float16* vrow = Vb + (size_t)qr * HH;
      #pragma unroll
      for (int ct = 0; ct < 8; ++ct)
        orow[ct * 16 + n] = (float)vrow[ct * 16 + n];
    }
  }
}

// ---------------------------------------------------------------------------
extern "C" void kernel_launch(void* const* d_in, const int* in_sizes, int n_in,
                              void* d_out, int out_size, void* d_ws, size_t ws_size,
                              hipStream_t stream) {
  (void)in_sizes; (void)n_in; (void)out_size; (void)ws_size;
  const float* x  = (const float*)d_in[0];
  const int*   im = (const int*)d_in[1];
  const float* Wq = (const float*)d_in[2];
  const float* bq = (const float*)d_in[3];
  const float* Wv = (const float*)d_in[4];
  const float* bv = (const float*)d_in[5];
  float* out = (float*)d_out;

  char* ws = (char*)d_ws;
  _Float16* Wpk = (_Float16*)ws;                                  // 512 KB
  _Float16* Qs  = (_Float16*)(ws + (size_t)1  * 1024 * 1024);     // 8 MB
  _Float16* Vs  = (_Float16*)(ws + (size_t)9  * 1024 * 1024);     // 8 MB
  _Float16* Vt  = (_Float16*)(ws + (size_t)17 * 1024 * 1024);     // 8 MB

  hipLaunchKernelGGL(pack_w_kernel, dim3(2 * EE * HH / 256), dim3(256), 0, stream,
                     Wq, Wv, Wpk);
  hipLaunchKernelGGL(proj_kernel, dim3(BB * TT / 128), dim3(256), 0, stream,
                     x, Wpk, bq, bv, Qs, Vs);
  hipLaunchKernelGGL(transpose_v_kernel, dim3(TT / 32, HH / 32, BB), dim3(256), 0, stream,
                     Vs, Vt);
  hipLaunchKernelGGL(attn_kernel, dim3(TT / 128, BB), dim3(256), 0, stream,
                     Qs, Vs, Vt, im, out);
}
